// EncoderLayer_22187801051722
// MI455X (gfx1250) — matrix operand order repaired and checked by reading, compiled, not run on hardware
//
#include <hip/hip_runtime.h>
#include <hip/hip_bf16.h>
#include <stdint.h>

typedef __attribute__((ext_vector_type(16))) _Float16 v16h;
typedef __attribute__((ext_vector_type(8)))  _Float16 h8;
typedef __attribute__((ext_vector_type(4)))  _Float16 h4;
typedef __attribute__((ext_vector_type(8)))  float    v8f;
typedef __attribute__((ext_vector_type(4)))  float    f4;

#define D_MODEL 1024
#define NUM_HEADS 16
#define HEAD_DIM 64
#define D_FF 4096
#define SEQ 2048
#define BATCH 2
#define ROWS (BATCH * SEQ)   /* 4096 */
#define EPS 1e-5f

// ---- helpers --------------------------------------------------------------

// Low 32 bits of a generic pointer to LDS == wave-relative LDS byte offset.
__device__ inline unsigned lds_off(const void* p) {
    return (unsigned)(uintptr_t)p;
}

// Per-lane 16-byte async copy global -> LDS (ASYNCcnt-tracked, no VGPR data).
__device__ inline void async_b128(const void* ldsp, const void* gp) {
    asm volatile("global_load_async_to_lds_b128 %0, %1, off"
                 :: "v"(lds_off(ldsp)), "v"((unsigned long long)(uintptr_t)gp)
                 : "memory");
}
__device__ inline void wait_async0() {
    asm volatile("s_wait_asynccnt 0" ::: "memory");
}

// Build a v16h WMMA fragment from two contiguous 16-byte runs in LDS.
__device__ inline v16h frag2(const _Float16* p0, const _Float16* p1) {
    h8 lo = *(const h8*)__builtin_assume_aligned(p0, 16);
    h8 hi = *(const h8*)__builtin_assume_aligned(p1, 16);
    v16h r;
#pragma unroll
    for (int e = 0; e < 8; ++e) { r[e] = lo[e]; r[e + 8] = hi[e]; }
    return r;
}

__device__ inline v8f wmma16(v16h a, v16h b, v8f c) {
    return __builtin_amdgcn_wmma_f32_16x16x32_f16(
        false, a, false, b, (short)0, c, false, false);
}

// ---- elementwise f32 -> f16 ----------------------------------------------
__global__ __launch_bounds__(256)
void cvt_f16_kernel(const float* __restrict__ in, _Float16* __restrict__ out, int n)
{
    int i = (blockIdx.x * 256 + threadIdx.x) * 4;
    if (i < n) {
        f4 v = *(const f4*)(in + i);
        h4 o;
        o[0] = (_Float16)v.x; o[1] = (_Float16)v.y;
        o[2] = (_Float16)v.z; o[3] = (_Float16)v.w;
        *(h4*)(out + i) = o;
    }
}

// ---- tiled transpose + convert: W[K][N] f32 -> Wt[N][K] f16 ---------------
__global__ __launch_bounds__(256)
void transpose_cvt_f16(const float* __restrict__ W, _Float16* __restrict__ Wt,
                       int K, int N)
{
    __shared__ float tile[32][33];
    const int t = threadIdx.x;
    const int r = t >> 3, c4 = (t & 7) * 4;
    const int kb = blockIdx.y * 32, nb = blockIdx.x * 32;
    f4 v = *(const f4*)(W + (size_t)(kb + r) * N + nb + c4);
    tile[r][c4 + 0] = v.x; tile[r][c4 + 1] = v.y;
    tile[r][c4 + 2] = v.z; tile[r][c4 + 3] = v.w;
    __syncthreads();
    _Float16* o = Wt + (size_t)(nb + r) * K + kb + c4;
    o[0] = (_Float16)tile[c4 + 0][r];
    o[1] = (_Float16)tile[c4 + 1][r];
    o[2] = (_Float16)tile[c4 + 2][r];
    o[3] = (_Float16)tile[c4 + 3][r];
}

// ---------------------------------------------------------------------------
// GEMM: out = act(Ah[M,Kd] @ Bt[N,Kd]^T + bias (+ resid)), f16 in, f32 accum.
// Block: 256 thr (8 waves), tile 64(M) x 128(N), K-step 32.
// Waves 2x4; each wave computes 32x32 = 2x2 WMMA tiles (register blocking).
// A/B staged with GLOBAL_LOAD_ASYNC_TO_LDS_B128; LDS row stride 48 halves
// (96 B, 16B-multiple) so fragment reads are aligned ds_load_b128.
// ---------------------------------------------------------------------------
__global__ __launch_bounds__(256)
void gemm_wmma_f16(const _Float16* __restrict__ Ah, const _Float16* __restrict__ Bt,
                   const float* __restrict__ bias, const float* __restrict__ resid,
                   float* __restrict__ out32, _Float16* __restrict__ out16,
                   int M, int N, int Kd, int relu)
{
    __shared__ __align__(16) _Float16 sA[64 * 48];    // [m][k]
    __shared__ __align__(16) _Float16 sB[128 * 48];   // [n][k]

    const int tid = threadIdx.x;
    const int lane = tid & 31, w = tid >> 5;
    const int lh = lane >> 4, lm = lane & 15;
    const int wm = w >> 2, wn = w & 3;                // 2 x 4 wave grid
    const int Mbase = blockIdx.y * 64;
    const int Nbase = blockIdx.x * 128;

    const int arow = tid >> 2, achk = tid & 3;        // A: 64 rows x 4 x 16B
    const int bn = tid >> 1, bch = (tid & 1) * 2;     // B: 128 rows x 4 x 16B, 2/thr

    v8f acc00 = {}, acc01 = {}, acc10 = {}, acc11 = {};

    for (int kb = 0; kb < Kd; kb += 32) {
        __syncthreads();   // prior iteration's fragment reads are complete
        const _Float16* ga = Ah + (size_t)(Mbase + arow) * Kd + kb + achk * 8;
        const _Float16* gb = Bt + (size_t)(Nbase + bn) * Kd + kb + bch * 8;
        async_b128(&sA[arow * 48 + achk * 8], ga);
        async_b128(&sB[bn * 48 + bch * 8], gb);
        async_b128(&sB[bn * 48 + (bch + 1) * 8], gb + 8);
        if (kb + 32 < Kd) {
            __builtin_prefetch(ga + 32, 0, 1);
            __builtin_prefetch(gb + 32, 0, 1);
        }
        wait_async0();
        __syncthreads();

        v16h a0 = frag2(&sA[(wm * 32 + lm) * 48 + 8 * lh],
                        &sA[(wm * 32 + lm) * 48 + 16 + 8 * lh]);
        v16h a1 = frag2(&sA[(wm * 32 + 16 + lm) * 48 + 8 * lh],
                        &sA[(wm * 32 + 16 + lm) * 48 + 16 + 8 * lh]);
        v16h b0 = frag2(&sB[(wn * 32 + lm) * 48 + 8 * lh],
                        &sB[(wn * 32 + lm) * 48 + 16 + 8 * lh]);
        v16h b1 = frag2(&sB[(wn * 32 + 16 + lm) * 48 + 8 * lh],
                        &sB[(wn * 32 + 16 + lm) * 48 + 16 + 8 * lh]);

        acc00 = wmma16(a0, b0, acc00);
        acc01 = wmma16(a0, b1, acc01);
        acc10 = wmma16(a1, b0, acc10);
        acc11 = wmma16(a1, b1, acc11);
    }

    // Epilogue. C layout: VGPR j -> row j + 8*lh (per 16-tile), col lm.
    const int r0 = Mbase + wm * 32 + lh * 8;
    const int r1 = r0 + 16;
    const int c0 = Nbase + wn * 32 + lm;
    const int c1 = c0 + 16;
    const float bv0 = bias ? bias[c0] : 0.0f;
    const float bv1 = bias ? bias[c1] : 0.0f;
#pragma unroll
    for (int j = 0; j < 8; ++j) {
        float v00 = acc00[j] + bv0, v01 = acc01[j] + bv1;
        float v10 = acc10[j] + bv0, v11 = acc11[j] + bv1;
        if (resid) {
            v00 += resid[(size_t)(r0 + j) * N + c0];
            v01 += resid[(size_t)(r0 + j) * N + c1];
            v10 += resid[(size_t)(r1 + j) * N + c0];
            v11 += resid[(size_t)(r1 + j) * N + c1];
        }
        if (relu) {
            v00 = fmaxf(v00, 0.0f); v01 = fmaxf(v01, 0.0f);
            v10 = fmaxf(v10, 0.0f); v11 = fmaxf(v11, 0.0f);
        }
        if (out32) {
            out32[(size_t)(r0 + j) * N + c0] = v00;
            out32[(size_t)(r0 + j) * N + c1] = v01;
            out32[(size_t)(r1 + j) * N + c0] = v10;
            out32[(size_t)(r1 + j) * N + c1] = v11;
        }
        if (out16) {
            out16[(size_t)(r0 + j) * N + c0] = (_Float16)v00;
            out16[(size_t)(r0 + j) * N + c1] = (_Float16)v01;
            out16[(size_t)(r1 + j) * N + c0] = (_Float16)v10;
            out16[(size_t)(r1 + j) * N + c1] = (_Float16)v11;
        }
    }
}

// ---------------------------------------------------------------------------
// Flash attention, f16 Q/K/V, f16 output. One block per (b*h, 128 q rows).
// K chunk staged via async b128; V transposed via VGPR scatter; P round-trips
// through wave-private LDS (LDS ops are in-order per wave).
// ---------------------------------------------------------------------------
__global__ __launch_bounds__(256)
void attention_flash(const _Float16* __restrict__ Q, const _Float16* __restrict__ Kh,
                     const _Float16* __restrict__ V, _Float16* __restrict__ O)
{
    __shared__ __align__(16) _Float16 sKt[32 * 72];      // [key][dim], 144B stride
    __shared__ __align__(16) _Float16 sVt[64 * 40];      // [dim][key]
    __shared__ __align__(16) _Float16 sP[8][16 * 40];    // per-wave P [row][k]

    const int tid = threadIdx.x;
    const int lane = tid & 31, w = tid >> 5;
    const int lh = lane >> 4, lm = lane & 15;
    const int bh = blockIdx.y;
    const int b = bh >> 4, h = bh & 15;
    const int qrow0 = b * SEQ + blockIdx.x * 128 + w * 16;
    const int kvbase = b * SEQ;
    const size_t hoff = (size_t)h * HEAD_DIM;

    v16h qf[2];
#pragma unroll
    for (int ks = 0; ks < 2; ++ks) {
#pragma unroll
        for (int e = 0; e < 16; ++e) {
            int kd = ks * 32 + ((e & 8) << 1) + (e & 7) + lh * 8;
            qf[ks][e] = Q[(size_t)(qrow0 + lm) * D_MODEL + hoff + kd];
        }
    }

    v8f o0 = {}, o1 = {}, o2 = {}, o3 = {};
    float mrow[8], lrow[8];
#pragma unroll
    for (int j = 0; j < 8; ++j) { mrow[j] = -3.0e38f; lrow[j] = 0.0f; }

    const float scale = 0.125f;
    const int kr = tid >> 3;                   // key 0..31
    const int dc8 = (tid & 7);                 // dim chunk (8 halves)

    for (int kc = 0; kc < SEQ; kc += 32) {
        __syncthreads();
        // K chunk: async 16B/lane into [key][dim]
        const _Float16* kp = Kh + (size_t)(kvbase + kc + kr) * D_MODEL + hoff + dc8 * 8;
        async_b128(&sKt[kr * 72 + dc8 * 8], kp);
        // V chunk: vector load + transpose scatter
        const _Float16* vp = V + (size_t)(kvbase + kc + kr) * D_MODEL + hoff + dc8 * 8;
        h8 vv = *(const h8*)__builtin_assume_aligned(vp, 16);
#pragma unroll
        for (int i = 0; i < 8; ++i) sVt[(dc8 * 8 + i) * 40 + kr] = vv[i];
        wait_async0();
        __syncthreads();

        // Scores: S[16q x 32k]
        v8f s0 = {}, s1 = {};
#pragma unroll
        for (int ks = 0; ks < 2; ++ks) {
            v16h bf0 = frag2(&sKt[lm * 72 + ks * 32 + 8 * lh],
                             &sKt[lm * 72 + ks * 32 + 16 + 8 * lh]);
            v16h bf1 = frag2(&sKt[(16 + lm) * 72 + ks * 32 + 8 * lh],
                             &sKt[(16 + lm) * 72 + ks * 32 + 16 + 8 * lh]);
            s0 = wmma16(qf[ks], bf0, s0);
            s1 = wmma16(qf[ks], bf1, s1);
        }

        // Online softmax per row (rows live across 16-lane half-groups)
#pragma unroll
        for (int j = 0; j < 8; ++j) {
            float x0 = s0[j] * scale, x1 = s1[j] * scale;
            float mx = fmaxf(x0, x1);
#pragma unroll
            for (int off = 1; off < 16; off <<= 1)
                mx = fmaxf(mx, __shfl_xor(mx, off, 32));
            float mnew = fmaxf(mrow[j], mx);
            float alpha = __expf(mrow[j] - mnew);
            float p0 = __expf(x0 - mnew), p1 = __expf(x1 - mnew);
            float ps = p0 + p1;
#pragma unroll
            for (int off = 1; off < 16; off <<= 1)
                ps += __shfl_xor(ps, off, 32);
            lrow[j] = lrow[j] * alpha + ps;
            mrow[j] = mnew;
            o0[j] *= alpha; o1[j] *= alpha; o2[j] *= alpha; o3[j] *= alpha;
            sP[w][(j + 8 * lh) * 40 + lm]      = (_Float16)p0;
            sP[w][(j + 8 * lh) * 40 + 16 + lm] = (_Float16)p1;
        }

        v16h pf = frag2(&sP[w][lm * 40 + 8 * lh],
                        &sP[w][lm * 40 + 16 + 8 * lh]);
#pragma unroll
        for (int n = 0; n < 4; ++n) {
            v16h vf = frag2(&sVt[(n * 16 + lm) * 40 + 8 * lh],
                            &sVt[(n * 16 + lm) * 40 + 16 + 8 * lh]);
            v8f* op = (n == 0) ? &o0 : (n == 1) ? &o1 : (n == 2) ? &o2 : &o3;
            *op = wmma16(pf, vf, *op);
        }
    }

    const int row0 = qrow0 + lh * 8;
#pragma unroll
    for (int j = 0; j < 8; ++j) {
        float inv = 1.0f / lrow[j];
        size_t base = (size_t)(row0 + j) * D_MODEL + hoff + lm;
        O[base + 0]  = (_Float16)(o0[j] * inv);
        O[base + 16] = (_Float16)(o1[j] * inv);
        O[base + 32] = (_Float16)(o2[j] * inv);
        O[base + 48] = (_Float16)(o3[j] * inv);
    }
}

// ---------------------------------------------------------------------------
// Row LayerNorm over D_MODEL=1024; optional f32 and f16 outputs.
// ---------------------------------------------------------------------------
__global__ __launch_bounds__(256)
void layernorm_row(const float* __restrict__ in, const float* __restrict__ g,
                   const float* __restrict__ be, float* __restrict__ out32,
                   _Float16* __restrict__ out16)
{
    __shared__ float red[256];
    const int row = blockIdx.x;
    const int tid = threadIdx.x;
    const float* r = in + (size_t)row * D_MODEL;

    float v[4];
    float s = 0.0f;
#pragma unroll
    for (int i = 0; i < 4; ++i) { v[i] = r[tid + i * 256]; s += v[i]; }
    red[tid] = s;
    __syncthreads();
    for (int st = 128; st > 0; st >>= 1) {
        if (tid < st) red[tid] += red[tid + st];
        __syncthreads();
    }
    const float mu = red[0] * (1.0f / D_MODEL);
    __syncthreads();

    float q = 0.0f;
#pragma unroll
    for (int i = 0; i < 4; ++i) { float d = v[i] - mu; q += d * d; }
    red[tid] = q;
    __syncthreads();
    for (int st = 128; st > 0; st >>= 1) {
        if (tid < st) red[tid] += red[tid + st];
        __syncthreads();
    }
    const float rstd = rsqrtf(red[0] * (1.0f / D_MODEL) + EPS);

#pragma unroll
    for (int i = 0; i < 4; ++i) {
        int c = tid + i * 256;
        float val = (v[i] - mu) * rstd * g[c] + be[c];
        if (out32) out32[(size_t)row * D_MODEL + c] = val;
        if (out16) out16[(size_t)row * D_MODEL + c] = (_Float16)val;
    }
}

// ---------------------------------------------------------------------------
extern "C" void kernel_launch(void* const* d_in, const int* in_sizes, int n_in,
                              void* d_out, int out_size, void* d_ws, size_t ws_size,
                              hipStream_t stream)
{
    (void)in_sizes; (void)n_in; (void)out_size; (void)ws_size;
    const float* x   = (const float*)d_in[0];
    const float* Wq  = (const float*)d_in[1];
    const float* bq  = (const float*)d_in[2];
    const float* Wk  = (const float*)d_in[3];
    const float* bk  = (const float*)d_in[4];
    const float* Wv  = (const float*)d_in[5];
    const float* bv  = (const float*)d_in[6];
    const float* Wo  = (const float*)d_in[7];
    const float* bo  = (const float*)d_in[8];
    const float* g1  = (const float*)d_in[9];
    const float* b1  = (const float*)d_in[10];
    const float* W1  = (const float*)d_in[11];
    const float* bf1 = (const float*)d_in[12];
    const float* W2  = (const float*)d_in[13];
    const float* bf2 = (const float*)d_in[14];
    const float* g2  = (const float*)d_in[15];
    const float* b2  = (const float*)d_in[16];
    float* out = (float*)d_out;

    const size_t RC = (size_t)ROWS * D_MODEL;           // 4M elements
    char* p = (char*)d_ws;
    auto take = [&](size_t bytes) {
        void* r = (void*)p;
        p += (bytes + 255) & ~(size_t)255;
        return r;
    };
    _Float16* xh   = (_Float16*)take(RC * 2);
    _Float16* Qh   = (_Float16*)take(RC * 2);
    _Float16* Kh   = (_Float16*)take(RC * 2);
    _Float16* Vh   = (_Float16*)take(RC * 2);
    _Float16* atth = (_Float16*)take(RC * 2);
    _Float16* hh   = (_Float16*)take(RC * 2);
    _Float16* ff1h = (_Float16*)take((size_t)ROWS * D_FF * 2);
    _Float16* Wqt  = (_Float16*)take((size_t)D_MODEL * D_MODEL * 2);
    _Float16* Wkt  = (_Float16*)take((size_t)D_MODEL * D_MODEL * 2);
    _Float16* Wvt  = (_Float16*)take((size_t)D_MODEL * D_MODEL * 2);
    _Float16* Wot  = (_Float16*)take((size_t)D_MODEL * D_MODEL * 2);
    _Float16* W1t  = (_Float16*)take((size_t)D_MODEL * D_FF * 2);
    _Float16* W2t  = (_Float16*)take((size_t)D_FF * D_MODEL * 2);
    float* y   = (float*)take(RC * 4);
    float* hb  = (float*)take(RC * 4);
    float* ff2 = (float*)take(RC * 4);

    dim3 blk(256);

    // Pre-convert activations & weights (weights transposed to [N][K] f16)
    cvt_f16_kernel<<<dim3((unsigned)(RC / 1024)), blk, 0, stream>>>(x, xh, (int)RC);
    dim3 gT(D_MODEL / 32, D_MODEL / 32);
    transpose_cvt_f16<<<gT, blk, 0, stream>>>(Wq, Wqt, D_MODEL, D_MODEL);
    transpose_cvt_f16<<<gT, blk, 0, stream>>>(Wk, Wkt, D_MODEL, D_MODEL);
    transpose_cvt_f16<<<gT, blk, 0, stream>>>(Wv, Wvt, D_MODEL, D_MODEL);
    transpose_cvt_f16<<<gT, blk, 0, stream>>>(Wo, Wot, D_MODEL, D_MODEL);
    transpose_cvt_f16<<<dim3(D_FF / 32, D_MODEL / 32), blk, 0, stream>>>(W1, W1t, D_MODEL, D_FF);
    transpose_cvt_f16<<<dim3(D_MODEL / 32, D_FF / 32), blk, 0, stream>>>(W2, W2t, D_FF, D_MODEL);

    dim3 gP(D_MODEL / 128, ROWS / 64);                  // 8 x 64
    gemm_wmma_f16<<<gP, blk, 0, stream>>>(xh, Wqt, bq, nullptr, nullptr, Qh,
                                          ROWS, D_MODEL, D_MODEL, 0);
    gemm_wmma_f16<<<gP, blk, 0, stream>>>(xh, Wkt, bk, nullptr, nullptr, Kh,
                                          ROWS, D_MODEL, D_MODEL, 0);
    gemm_wmma_f16<<<gP, blk, 0, stream>>>(xh, Wvt, bv, nullptr, nullptr, Vh,
                                          ROWS, D_MODEL, D_MODEL, 0);

    attention_flash<<<dim3(SEQ / 128, BATCH * NUM_HEADS), blk, 0, stream>>>(Qh, Kh, Vh, atth);

    gemm_wmma_f16<<<gP, blk, 0, stream>>>(atth, Wot, bo, x, y, nullptr,
                                          ROWS, D_MODEL, D_MODEL, 0);
    layernorm_row<<<dim3(ROWS), blk, 0, stream>>>(y, g1, b1, hb, hh);

    gemm_wmma_f16<<<dim3(D_FF / 128, ROWS / 64), blk, 0, stream>>>(
        hh, W1t, bf1, nullptr, nullptr, ff1h, ROWS, D_FF, D_MODEL, 1);
    gemm_wmma_f16<<<gP, blk, 0, stream>>>(ff1h, W2t, bf2, hb, ff2, nullptr,
                                          ROWS, D_MODEL, D_FF, 0);

    layernorm_row<<<dim3(ROWS), blk, 0, stream>>>(ff2, g2, b2, out, nullptr);
}
